// I2P_CrossAttention_54296976556403
// MI455X (gfx1250) — matrix operand order repaired and checked by reading, not compiled, not run on hardware
//
#include <hip/hip_runtime.h>
#include <hip/hip_bf16.h>

// Problem dimensions (fixed by the reference)
#define IN_C   512
#define DMODEL 512
#define NQ     2048
#define MKV    4096
#define NBATCH 4

typedef _Float16 f16;
typedef __attribute__((ext_vector_type(16))) _Float16 v16h;
typedef __attribute__((ext_vector_type(8)))  _Float16 v8h;
typedef __attribute__((ext_vector_type(4)))  _Float16 f16x4;
typedef __attribute__((ext_vector_type(8)))  float    v8f;
typedef __attribute__((ext_vector_type(4)))  unsigned int u32x4;
typedef __attribute__((ext_vector_type(4)))  int      i32x4;
typedef __attribute__((ext_vector_type(8)))  int      i32x8;

union Frag32B { v16h h; u32x4 q[2]; };

__device__ __forceinline__ v8f wmma16x16x32(v16h a, v16h b, v8f c) {
    // D = A(16x32 f16) * B(32x16 f16) + C(16x16 f32)
    return __builtin_amdgcn_wmma_f32_16x16x32_f16(
        /*neg_a=*/false, a, /*neg_b=*/false, b,
        /*c_mod=*/(short)0, c, /*reuse_a=*/false, /*reuse_b=*/false);
}

// ---------------------------------------------------------------------------
// Kernel 0: f32 -> f16 weight conversion (vectorized x4)
// ---------------------------------------------------------------------------
__global__ void __launch_bounds__(256)
cvt_f32_f16(const float* __restrict__ src, f16* __restrict__ dst, int n) {
    int i = (blockIdx.x * 256 + threadIdx.x) * 4;
    if (i + 3 < n) {
        float4 v = *(const float4*)(src + i);
        f16x4 h;
        h[0] = (f16)v.x; h[1] = (f16)v.y; h[2] = (f16)v.z; h[3] = (f16)v.w;
        *(f16x4*)(dst + i) = h;
    }
}

// ---------------------------------------------------------------------------
// Kernel 1: out[b,i,d] = sum_c x[b,c,i] * W[d,c] + bias[d]   (f16 out)
// x is [B,C,I] (i contiguous) -> A tile transposed via LDS with f32->f16 cvt.
// W is f16 [D,C] (c contiguous) -> B fragments loaded straight from global.
// STORET==1 stores out transposed as [B,D,I] (i contiguous) for V.
// Workgroup: 256 threads = 8 waves; tile 128(i) x 64(d); wave = 32x32.
// I and STORET are compile-time so the backend strength-reduces addressing.
// ---------------------------------------------------------------------------
template <int I, int STORET>
__global__ void __launch_bounds__(256)
qkv_gemm(const float* __restrict__ x, const f16* __restrict__ Wh,
         const float* __restrict__ bias, f16* __restrict__ out)
{
    __shared__ __align__(16) f16 lA[128][40];   // [i][c], padded rows (80B)
    const int b   = blockIdx.z;
    const int i0  = blockIdx.x * 128;
    const int d0  = blockIdx.y * 64;
    const int tid = threadIdx.x;
    const int lane = tid & 31;
    const int wid  = tid >> 5;
    const int wi = wid & 3, wd = wid >> 2;      // 4 waves along i, 2 along d
    const int ln = lane & 15;
    const int lh = lane >> 4;
    const int khalf = lane & 16;
    const float* xb = x + (size_t)b * IN_C * I;

    v8f zero = {0.f,0.f,0.f,0.f,0.f,0.f,0.f,0.f};
    v8f acc[2][2];
#pragma unroll
    for (int a = 0; a < 2; ++a)
#pragma unroll
        for (int c = 0; c < 2; ++c) acc[a][c] = zero;

    const int tr = tid >> 5;            // c row 0..7 (+8 per rep)
    const int tc = (tid & 31) * 4;      // i col 0..124

    for (int c0 = 0; c0 < IN_C; c0 += 32) {
        // ---- stage A tile: 32 c-rows x 128 i-cols, transpose into LDS ----
#pragma unroll
        for (int rep = 0; rep < 4; ++rep) {
            int cc = tr + rep * 8;
            float4 v = *(const float4*)(xb + (size_t)(c0 + cc) * I + i0 + tc);
            lA[tc + 0][cc] = (f16)v.x;
            lA[tc + 1][cc] = (f16)v.y;
            lA[tc + 2][cc] = (f16)v.z;
            lA[tc + 3][cc] = (f16)v.w;
        }
        __syncthreads();

        // ---- fragments ----
        Frag32B af[2], bf[2];
#pragma unroll
        for (int fi = 0; fi < 2; ++fi) {
            const f16* p = &lA[wi * 32 + fi * 16 + ln][khalf ? 16 : 0];
            af[fi].q[0] = *(const u32x4*)p;
            af[fi].q[1] = *(const u32x4*)(p + 8);
        }
#pragma unroll
        for (int fd = 0; fd < 2; ++fd) {
            int d = d0 + wd * 32 + fd * 16 + ln;
            const f16* p = Wh + (size_t)d * IN_C + c0 + (khalf ? 16 : 0);
            bf[fd].q[0] = *(const u32x4*)p;
            bf[fd].q[1] = *(const u32x4*)(p + 8);
        }
#pragma unroll
        for (int fi = 0; fi < 2; ++fi)
#pragma unroll
            for (int fd = 0; fd < 2; ++fd)
                acc[fi][fd] = wmma16x16x32(af[fi].h, bf[fd].h, acc[fi][fd]);
        __syncthreads();
    }

    // ---- epilogue: bias + store f16 ----
#pragma unroll
    for (int fi = 0; fi < 2; ++fi) {
        int ibase = i0 + wi * 32 + fi * 16 + 8 * lh;
#pragma unroll
        for (int fd = 0; fd < 2; ++fd) {
            int d = d0 + wd * 32 + fd * 16 + ln;
            float bv = bias[d];
            if (STORET) {   // [B,D,I]: rows (g) consecutive in i -> one b128
                v8h pk;
#pragma unroll
                for (int g = 0; g < 8; ++g) pk[g] = (f16)(acc[fi][fd][g] + bv);
                *(v8h*)(out + ((size_t)b * DMODEL + d) * I + ibase) = pk;
            } else {        // [B,I,D]
#pragma unroll
                for (int g = 0; g < 8; ++g)
                    out[((size_t)b * I + ibase + g) * DMODEL + d] =
                        (f16)(acc[fi][fd][g] + bv);
            }
        }
    }
}

// ---------------------------------------------------------------------------
// Kernel 2: fused flash attention.
// ctx[b,n,d] = (softmax_m(q.k^T)/sqrt(C)) @ v    (f16 out)
// Workgroup: 128 threads = 4 waves, 16 query rows per wave (64 per WG).
// Q tile staged in LDS via the Tensor Data Mover (TDM) with LDS row padding
// expressed through the D# pad fields; K/V fragments direct from global.
// ---------------------------------------------------------------------------
__global__ void __launch_bounds__(128)
flash_attn(const f16* __restrict__ Qh, const f16* __restrict__ Kh,
           const f16* __restrict__ Vt, f16* __restrict__ ctxh)
{
    __shared__ __align__(16) f16 Qs[64 * 520];      // 64 rows, 1024B+16B pad
    __shared__ __align__(16) f16 Pt[4][16][32];     // per-wave P tile
    const int b   = blockIdx.y;
    const int tid = threadIdx.x;
    const int lane = tid & 31, wid = tid >> 5;
    const int ln = lane & 15, lh = lane >> 4, khalf = lane & 16;
    const int n0wg = blockIdx.x * 64;
    const f16* Qb = Qh + ((size_t)b * NQ + n0wg) * DMODEL;

    // ---- stage Q tile (64 x 512 f16) into LDS ----
#if __has_builtin(__builtin_amdgcn_tensor_load_to_lds)
    if (tid < 32) {   // one wave issues the TDM transfer (EXEC ignored by TDM)
        unsigned long long ga = (unsigned long long)(size_t)Qb;
        unsigned int ldsoff = (unsigned int)(unsigned long long)(size_t)(&Qs[0]);
        u32x4 g0;
        g0[0] = 1u;                                      // count=1, user mode
        g0[1] = ldsoff;                                  // lds_addr (bytes)
        g0[2] = (unsigned int)(ga & 0xffffffffu);        // global_addr[31:0]
        g0[3] = (unsigned int)((ga >> 32) & 0x01ffffffu) // global_addr[56:32]
                | 0x80000000u;                           // type = 2 ("image")
        i32x8 g1;
        g1[0] = (int)((1u << 16)      // data_size = 2 bytes
                    | (1u << 20)      // pad_enable
                    | (7u << 22)      // pad_interval: every 256 DWORDs (1024B)
                    | (3u << 25));    // pad_amount: 4 DWORDs (16B)
        g1[1] = (int)(512u << 16);    // tensor_dim0 = 512 (bits [79:48] lo)
        g1[2] = (int)(64u  << 16);    // dim0 hi=0 | tensor_dim1 = 64
        g1[3] = (int)(512u << 16);    // dim1 hi=0 | tile_dim0 = 512
        g1[4] = 64;                   // tile_dim1 = 64, tile_dim2 = 0
        g1[5] = 512;                  // tensor_dim0_stride = 512 (lo32)
        g1[6] = 0;                    // stride hi | tensor_dim1_stride lo
        g1[7] = 0;
        i32x4 gz = {0, 0, 0, 0};
#if __clang_major__ >= 23
        i32x8 gz8 = {0, 0, 0, 0, 0, 0, 0, 0};
        __builtin_amdgcn_tensor_load_to_lds(g0, g1, gz, gz, gz8, 0);
#else
        __builtin_amdgcn_tensor_load_to_lds(g0, g1, gz, gz, 0);
#endif
#if __has_builtin(__builtin_amdgcn_s_wait_tensorcnt)
        __builtin_amdgcn_s_wait_tensorcnt(0);
#else
        asm volatile("s_wait_tensorcnt 0x0" ::: "memory");
#endif
    }
#else
    {
#pragma unroll
        for (int it = 0; it < 32; ++it) {
            int chunk = tid + 128 * it;          // 0..4095 (8-half chunks)
            int row = chunk >> 6;
            int col = (chunk & 63) * 8;
            *(u32x4*)(Qs + row * 520 + col) =
                *(const u32x4*)(Qb + (size_t)row * DMODEL + col);
        }
    }
#endif
    __syncthreads();

    const int nw = wid * 16;                     // wave row base in Qs
    v8f zero = {0.f,0.f,0.f,0.f,0.f,0.f,0.f,0.f};
    v8f acc[32];                                 // ctx accum: 32 d-frags
#pragma unroll
    for (int f = 0; f < 32; ++f) acc[f] = zero;
    float mrow[8], lsum[8];
#pragma unroll
    for (int g = 0; g < 8; ++g) { mrow[g] = -3.0e38f; lsum[g] = 0.f; }

    const f16* Kb = Kh + (size_t)b * MKV * DMODEL;
    const f16* Vb = Vt + (size_t)b * DMODEL * MKV;

    for (int mt = 0; mt < MKV; mt += 32) {
        // prefetch next K tile rows into the cache hierarchy
        if (mt + 32 < MKV)
            __builtin_prefetch(Kb + (size_t)(mt + 32 + lane) * DMODEL, 0, 0);

        // ---- S = Q.K^T for 32 key columns ----
        v8f s0 = zero, s1 = zero;
#pragma unroll
        for (int kc = 0; kc < 16; ++kc) {
            Frag32B af;
            const f16* ap = Qs + (nw + ln) * 520 + kc * 32 + (khalf ? 16 : 0);
            af.q[0] = *(const u32x4*)ap;
            af.q[1] = *(const u32x4*)(ap + 8);
            const f16* kp0 = Kb + (size_t)(mt + ln) * DMODEL + kc * 32 + (khalf ? 16 : 0);
            Frag32B bf0; bf0.q[0] = *(const u32x4*)kp0; bf0.q[1] = *(const u32x4*)(kp0 + 8);
            s0 = wmma16x16x32(af.h, bf0.h, s0);
            const f16* kp1 = kp0 + (size_t)16 * DMODEL;
            Frag32B bf1; bf1.q[0] = *(const u32x4*)kp1; bf1.q[1] = *(const u32x4*)(kp1 + 8);
            s1 = wmma16x16x32(af.h, bf1.h, s1);
        }

        // ---- online softmax: row r = g + 8*lh, cols across 16 lanes ----
        float scale[8];
#pragma unroll
        for (int g = 0; g < 8; ++g) {
            float cmax = fmaxf(s0[g], s1[g]);
#pragma unroll
            for (int off = 8; off >= 1; off >>= 1)
                cmax = fmaxf(cmax, __shfl_xor(cmax, off, 16));
            float nm = fmaxf(mrow[g], cmax);
            float p0 = __expf(s0[g] - nm);
            float p1 = __expf(s1[g] - nm);
            float rs = p0 + p1;
#pragma unroll
            for (int off = 8; off >= 1; off >>= 1)
                rs += __shfl_xor(rs, off, 16);
            scale[g] = __expf(mrow[g] - nm);
            lsum[g]  = lsum[g] * scale[g] + rs;
            mrow[g]  = nm;
            s0[g] = p0; s1[g] = p1;
        }
#pragma unroll
        for (int f = 0; f < 32; ++f)
#pragma unroll
            for (int g = 0; g < 8; ++g) acc[f][g] *= scale[g];

        // ---- P: D-layout f32 -> A-layout f16 via per-wave LDS tile ----
#pragma unroll
        for (int g = 0; g < 8; ++g) {
            Pt[wid][8 * lh + g][ln]      = (f16)s0[g];
            Pt[wid][8 * lh + g][16 + ln] = (f16)s1[g];
        }
        Frag32B pf;
        const f16* pp = &Pt[wid][ln][khalf ? 16 : 0];
        pf.q[0] = *(const u32x4*)pp;
        pf.q[1] = *(const u32x4*)(pp + 8);

        // ---- ctx += P @ V ----
#pragma unroll
        for (int f = 0; f < 32; ++f) {
            const f16* vp = Vb + (size_t)(f * 16 + ln) * MKV + mt + (khalf ? 16 : 0);
            Frag32B vf; vf.q[0] = *(const u32x4*)vp; vf.q[1] = *(const u32x4*)(vp + 8);
            acc[f] = wmma16x16x32(pf.h, vf.h, acc[f]);
        }
    }

    // ---- finalize: divide by row sum and by sqrt(C); store f16 ----
    const float rsc = 0.044194173824159216f;     // 1/sqrt(512)
    float inv[8];
#pragma unroll
    for (int g = 0; g < 8; ++g) inv[g] = rsc / lsum[g];
#pragma unroll
    for (int f = 0; f < 32; ++f) {
        int d = f * 16 + ln;
#pragma unroll
        for (int g = 0; g < 8; ++g) {
            int n = n0wg + nw + 8 * lh + g;
            ctxh[((size_t)b * NQ + n) * DMODEL + d] = (f16)(acc[f][g] * inv[g]);
        }
    }
}

// ---------------------------------------------------------------------------
// Kernel 3: out[b,e,n] = feat0[b,e,n] + bo[e] + sum_d ctx[b,n,d]*Wo[e,d]
// Fused output GEMM + bias + residual + transpose. Two e-passes of 16 frags.
// ---------------------------------------------------------------------------
__global__ void __launch_bounds__(128)
out_proj(const f16* __restrict__ ctxh, const f16* __restrict__ Woh,
         const float* __restrict__ bo, const float* __restrict__ feat0,
         float* __restrict__ out)
{
    const int b   = blockIdx.y;
    const int tid = threadIdx.x;
    const int lane = tid & 31, wid = tid >> 5;
    const int ln = lane & 15, lh = lane >> 4, khalf = lane & 16;
    const int n0 = blockIdx.x * 64 + wid * 16;
    const f16* cb = ctxh + ((size_t)b * NQ + n0) * DMODEL;
    v8f zero = {0.f,0.f,0.f,0.f,0.f,0.f,0.f,0.f};

    for (int pass = 0; pass < 2; ++pass) {
        v8f acc[16];
#pragma unroll
        for (int ef = 0; ef < 16; ++ef) acc[ef] = zero;

        for (int kc = 0; kc < 16; ++kc) {
            Frag32B af;
            const f16* ap = cb + (size_t)ln * DMODEL + kc * 32 + (khalf ? 16 : 0);
            af.q[0] = *(const u32x4*)ap;
            af.q[1] = *(const u32x4*)(ap + 8);
#pragma unroll
            for (int ef = 0; ef < 16; ++ef) {
                int e = pass * 256 + ef * 16 + ln;
                const f16* wp = Woh + (size_t)e * DMODEL + kc * 32 + (khalf ? 16 : 0);
                Frag32B bf; bf.q[0] = *(const u32x4*)wp; bf.q[1] = *(const u32x4*)(wp + 8);
                acc[ef] = wmma16x16x32(af.h, bf.h, acc[ef]);
            }
        }

        // epilogue: rows g=0..7 are consecutive n -> 2x float4 load/store
#pragma unroll
        for (int ef = 0; ef < 16; ++ef) {
            int e = pass * 256 + ef * 16 + ln;
            float be = bo[e];
            size_t base = ((size_t)b * DMODEL + e) * NQ + n0 + 8 * lh;
            float4 r0 = *(const float4*)(feat0 + base);
            float4 r1 = *(const float4*)(feat0 + base + 4);
            float4 o0, o1;
            o0.x = acc[ef][0] + be + r0.x;
            o0.y = acc[ef][1] + be + r0.y;
            o0.z = acc[ef][2] + be + r0.z;
            o0.w = acc[ef][3] + be + r0.w;
            o1.x = acc[ef][4] + be + r1.x;
            o1.y = acc[ef][5] + be + r1.y;
            o1.z = acc[ef][6] + be + r1.z;
            o1.w = acc[ef][7] + be + r1.w;
            *(float4*)(out + base)     = o0;
            *(float4*)(out + base + 4) = o1;
        }
    }
}

// ---------------------------------------------------------------------------
// Launch
// ---------------------------------------------------------------------------
extern "C" void kernel_launch(void* const* d_in, const int* in_sizes, int n_in,
                              void* d_out, int out_size, void* d_ws, size_t ws_size,
                              hipStream_t stream) {
    const float* feat0 = (const float*)d_in[0];   // [B,C,N]
    const float* feat1 = (const float*)d_in[1];   // [B,C,W,H] = [B,C,M]
    const float* Wq = (const float*)d_in[2];
    const float* bq = (const float*)d_in[3];
    const float* Wk = (const float*)d_in[4];
    const float* bk = (const float*)d_in[5];
    const float* Wv = (const float*)d_in[6];
    const float* bv = (const float*)d_in[7];
    const float* Wo = (const float*)d_in[8];
    const float* bo = (const float*)d_in[9];
    float* out = (float*)d_out;

    char* ws = (char*)d_ws;
    size_t off = 0;
    const size_t WBYTES = (size_t)DMODEL * IN_C * sizeof(f16);      // 512KB
    f16* Wqh = (f16*)(ws + off); off += WBYTES;
    f16* Wkh = (f16*)(ws + off); off += WBYTES;
    f16* Wvh = (f16*)(ws + off); off += WBYTES;
    f16* Woh = (f16*)(ws + off); off += WBYTES;
    f16* Qh  = (f16*)(ws + off); off += (size_t)NBATCH * NQ  * DMODEL * sizeof(f16);
    f16* Kh  = (f16*)(ws + off); off += (size_t)NBATCH * MKV * DMODEL * sizeof(f16);
    f16* Vt  = (f16*)(ws + off); off += (size_t)NBATCH * DMODEL * MKV * sizeof(f16);  // [B,D,M]
    f16* ctxh= (f16*)(ws + off); off += (size_t)NBATCH * NQ  * DMODEL * sizeof(f16);
    (void)off; (void)in_sizes; (void)n_in; (void)out_size; (void)ws_size;

    const int nW = DMODEL * IN_C;                 // 262144
    const int cvtBlocks = nW / (256 * 4);
    cvt_f32_f16<<<cvtBlocks, 256, 0, stream>>>(Wq, Wqh, nW);
    cvt_f32_f16<<<cvtBlocks, 256, 0, stream>>>(Wk, Wkh, nW);
    cvt_f32_f16<<<cvtBlocks, 256, 0, stream>>>(Wv, Wvh, nW);
    cvt_f32_f16<<<cvtBlocks, 256, 0, stream>>>(Wo, Woh, nW);

    qkv_gemm<NQ, 0><<<dim3(NQ / 128,  DMODEL / 64, NBATCH), 256, 0, stream>>>(feat0, Wqh, bq, Qh);
    qkv_gemm<MKV, 0><<<dim3(MKV / 128, DMODEL / 64, NBATCH), 256, 0, stream>>>(feat1, Wkh, bk, Kh);
    qkv_gemm<MKV, 1><<<dim3(MKV / 128, DMODEL / 64, NBATCH), 256, 0, stream>>>(feat1, Wvh, bv, Vt);

    flash_attn<<<dim3(NQ / 64, NBATCH), 128, 0, stream>>>(Qh, Kh, Vt, ctxh);

    out_proj<<<dim3(NQ / 64, NBATCH), 128, 0, stream>>>(ctxh, Woh, bo, feat0, out);
}
